// G_25177098289494
// MI455X (gfx1250) — matrix operand order repaired and checked by reading, not compiled, not run on hardware
//
#include <hip/hip_runtime.h>
#include <hip/hip_bf16.h>

// ---------------- problem constants ----------------
static constexpr int B    = 32768;
static constexpr int RNN  = 128;
static constexpr int AUX  = 32;
static constexpr int FEAT = 24;
static constexpr int FC   = 128;
static constexpr int NC   = 256;
static constexpr int K0   = 64;   // 1+24+32=57 padded to 64
static constexpr int G3   = 3 * RNN;  // 384

typedef __bf16 bf16_t;
typedef __attribute__((ext_vector_type(16))) __bf16 bf16x16;
typedef __attribute__((ext_vector_type(8)))  __bf16 bf16x8;
typedef __attribute__((ext_vector_type(8)))  float  f32x8;

// ---------------- prep kernels ----------------

// weight f32 [rows, scols] -> bf16 [rows, dcols] (zero pad cols >= scols)
__global__ void k_w2bf16(bf16_t* __restrict__ dst, const float* __restrict__ src,
                         int rows, int scols, int dcols) {
  int t = blockIdx.x * blockDim.x + threadIdx.x;
  if (t >= rows * dcols) return;
  int r = t / dcols, c = t - r * dcols;
  dst[t] = (c < scols) ? (bf16_t)src[r * scols + c] : (bf16_t)0.0f;
}

// contiguous f32 -> bf16
__global__ void k_f2b(bf16_t* __restrict__ dst, const float* __restrict__ src, int n) {
  int t = blockIdx.x * blockDim.x + threadIdx.x;
  if (t < n) dst[t] = (bf16_t)src[t];
}

// X0[row, 0:64] = [x, m_t(24), a1_t(32), 0-pad(7)]  (bf16)
__global__ void k_pack_x0(bf16_t* __restrict__ X0, const float* __restrict__ x,
                          const float* __restrict__ m, const float* __restrict__ a1) {
  int row = blockIdx.x;
  int c = threadIdx.x;               // blockDim = 64
  float v = 0.0f;
  if (c == 0)       v = x[row];
  else if (c < 25)  v = m[(size_t)row * FEAT + (c - 1)];
  else if (c < 57)  v = a1[(size_t)row * AUX + (c - 25)];
  X0[(size_t)row * K0 + c] = (bf16_t)v;
}

// dst[row, 128+c] = bf16(aux[row, c])   (dst ld = 160)
__global__ void k_aux32(bf16_t* __restrict__ dst, const float* __restrict__ aux) {
  int row = blockIdx.x;
  int c = threadIdx.x;               // blockDim = 32
  dst[(size_t)row * (RNN + AUX) + RNN + c] = (bf16_t)aux[(size_t)row * AUX + c];
}

// ---------------- WMMA GEMM ----------------
// Y[M,N] = act(X[M,K](bf16, ld=K) @ W[N,K](bf16, ld=K)^T + bias[N])
// grid: (M/16, N/128); block: 256 (8 waves, one 16x16 tile per wave)
__global__ void k_wmma_gemm(const bf16_t* __restrict__ X, const bf16_t* __restrict__ W,
                            const float* __restrict__ bias,
                            float* __restrict__ Yf, int ldyf,
                            bf16_t* __restrict__ Yb, int ldyb,
                            int N, int K, int relu) {
  const int wave  = threadIdx.x >> 5;
  const int lane  = threadIdx.x & 31;
  const int l15   = lane & 15;
  const int khalf = lane >> 4;       // 0 or 1
  const int m_base = blockIdx.x * 16;
  const int n_base = (blockIdx.y * 8 + wave) * 16;
  if (n_base >= N) return;           // wave-uniform; EXEC stays all-1s for WMMA

  const bf16_t* arow = X + (size_t)(m_base + l15) * K + khalf * 16;
  const bf16_t* brow = W + (size_t)(n_base + l15) * K + khalf * 16;

  f32x8 c = {};
  for (int k0 = 0; k0 < K; k0 += 32) {
    union { bf16x16 v; bf16x8 h[2]; } a;
    a.h[0] = *(const bf16x8*)(arow + k0);        // K = k0 + khalf*16 + [0..7]
    a.h[1] = *(const bf16x8*)(arow + k0 + 8);    // K = k0+8 + khalf*16 + [0..7]
    bf16x16 b = *(const bf16x16*)(brow + k0);    // K = k0 + khalf*16 + [0..15]
    c = __builtin_amdgcn_wmma_f32_16x16x32_bf16(
        /*neg_a=*/false, a.v, /*neg_b=*/false, b,
        /*c_mod=*/(short)0, c, /*reuse_a=*/false, /*reuse_b=*/false);
  }

  const int n = n_base + l15;
  const float bv = bias[n];
#pragma unroll
  for (int j = 0; j < 8; ++j) {
    int m = m_base + khalf * 8 + j;
    float v = c[j] + bv;
    if (relu) v = fmaxf(v, 0.0f);
    if (Yf) Yf[(size_t)m * ldyf + n] = v;
    if (Yb) Yb[(size_t)m * ldyb + n] = (bf16_t)v;
  }
}

// ---------------- GRU gate combine ----------------
// h' = (1-z)*n + z*h ; s = h' + res ; xnext[:,0:128]=bf16(s); xnext[:,128:160]=bf16(aux)
// grid: B blocks; block: 160 threads (5 waves)
__global__ void k_gru_combine(const float* __restrict__ gi, const float* __restrict__ gh,
                              const float* __restrict__ hprev, const float* __restrict__ res,
                              const float* __restrict__ aux,
                              float* __restrict__ hout, float* __restrict__ sumf,
                              bf16_t* __restrict__ xnext) {
  const int row = blockIdx.x;
  const int j = threadIdx.x;
  if (j < RNN) {
    const float* girow = gi + (size_t)row * G3;
    const float* ghrow = gh + (size_t)row * G3;
    float r = 1.0f / (1.0f + __expf(-(girow[j]         + ghrow[j])));
    float z = 1.0f / (1.0f + __expf(-(girow[RNN + j]   + ghrow[RNN + j])));
    float n = tanhf(girow[2 * RNN + j] + r * ghrow[2 * RNN + j]);
    float hp = hprev[(size_t)row * RNN + j];
    float h  = (1.0f - z) * n + z * hp;
    hout[(size_t)row * RNN + j] = h;
    float s = h + res[(size_t)row * RNN + j];
    if (sumf) sumf[(size_t)row * RNN + j] = s;
    xnext[(size_t)row * (RNN + AUX) + j] = (bf16_t)s;
  } else {
    int a = j - RNN;
    xnext[(size_t)row * (RNN + AUX) + j] = (bf16_t)aux[(size_t)row * AUX + a];
  }
}

// ---------------- host ----------------
extern "C" void kernel_launch(void* const* d_in, const int* in_sizes, int n_in,
                              void* d_out, int out_size, void* d_ws, size_t ws_size,
                              hipStream_t stream) {
  (void)in_sizes; (void)n_in; (void)out_size; (void)ws_size;
  const float* x      = (const float*)d_in[0];
  const float* h1     = (const float*)d_in[1];
  const float* h2     = (const float*)d_in[2];
  const float* m_t    = (const float*)d_in[3];
  const float* a1     = (const float*)d_in[4];
  const float* a2     = (const float*)d_in[5];
  const float* a3     = (const float*)d_in[6];
  const float* a4     = (const float*)d_in[7];
  const float* I_w    = (const float*)d_in[8];
  const float* I_b    = (const float*)d_in[9];
  const float* r1_wih = (const float*)d_in[10];
  const float* r1_whh = (const float*)d_in[11];
  const float* r1_bih = (const float*)d_in[12];
  const float* r1_bhh = (const float*)d_in[13];
  const float* r2_wih = (const float*)d_in[14];
  const float* r2_whh = (const float*)d_in[15];
  const float* r2_bih = (const float*)d_in[16];
  const float* r2_bhh = (const float*)d_in[17];
  const float* fc1_w  = (const float*)d_in[18];
  const float* fc1_b  = (const float*)d_in[19];
  const float* fc2_w  = (const float*)d_in[20];
  const float* fc2_b  = (const float*)d_in[21];
  const float* fc3_w  = (const float*)d_in[22];
  const float* fc3_b  = (const float*)d_in[23];

  float* logits = (float*)d_out;                       // [B, 256]
  float* h1o    = logits + (size_t)B * NC;             // [B, 128]
  float* h2o    = h1o + (size_t)B * RNN;               // [B, 128]

  char* ws = (char*)d_ws;
  size_t off = 0;
  auto alloc = [&](size_t bytes) -> char* {
    char* p = ws + off;
    off = (off + bytes + 255) & ~(size_t)255;
    return p;
  };

  // bf16 weights
  bf16_t* wI   = (bf16_t*)alloc((size_t)RNN * K0 * 2);
  bf16_t* w1ih = (bf16_t*)alloc((size_t)G3 * RNN * 2);
  bf16_t* w1hh = (bf16_t*)alloc((size_t)G3 * RNN * 2);
  bf16_t* w2ih = (bf16_t*)alloc((size_t)G3 * (RNN + AUX) * 2);
  bf16_t* w2hh = (bf16_t*)alloc((size_t)G3 * RNN * 2);
  bf16_t* wf1  = (bf16_t*)alloc((size_t)FC * (RNN + AUX) * 2);
  bf16_t* wf2  = (bf16_t*)alloc((size_t)FC * (FC + AUX) * 2);
  bf16_t* wf3  = (bf16_t*)alloc((size_t)NC * FC * 2);
  // activations
  bf16_t* X0    = (bf16_t*)alloc((size_t)B * K0 * 2);
  bf16_t* H1b   = (bf16_t*)alloc((size_t)B * RNN * 2);
  bf16_t* H2b   = (bf16_t*)alloc((size_t)B * RNN * 2);
  float*  io_f  = (float*) alloc((size_t)B * RNN * 4);
  bf16_t* io_b  = (bf16_t*)alloc((size_t)B * RNN * 2);
  float*  gi    = (float*) alloc((size_t)B * G3 * 4);
  float*  gh    = (float*) alloc((size_t)B * G3 * 4);
  float*  g2in  = (float*) alloc((size_t)B * RNN * 4);
  bf16_t* X2    = (bf16_t*)alloc((size_t)B * (RNN + AUX) * 2);
  bf16_t* X3    = (bf16_t*)alloc((size_t)B * (RNN + AUX) * 2);
  bf16_t* X4    = (bf16_t*)alloc((size_t)B * (FC + AUX) * 2);
  bf16_t* R2b   = (bf16_t*)alloc((size_t)B * FC * 2);

  auto cvt = [&](bf16_t* dst, const float* src, int rows, int scols, int dcols) {
    int total = rows * dcols;
    k_w2bf16<<<(total + 255) / 256, 256, 0, stream>>>(dst, src, rows, scols, dcols);
  };
  // weight conversion (I_w K-padded 57 -> 64)
  cvt(wI,   I_w,    RNN, FEAT + AUX + 1, K0);
  cvt(w1ih, r1_wih, G3,  RNN,       RNN);
  cvt(w1hh, r1_whh, G3,  RNN,       RNN);
  cvt(w2ih, r2_wih, G3,  RNN + AUX, RNN + AUX);
  cvt(w2hh, r2_whh, G3,  RNN,       RNN);
  cvt(wf1,  fc1_w,  FC,  RNN + AUX, RNN + AUX);
  cvt(wf2,  fc2_w,  FC,  FC + AUX,  FC + AUX);
  cvt(wf3,  fc3_w,  NC,  FC,        FC);

  // input packing
  k_pack_x0<<<B, K0, 0, stream>>>(X0, x, m_t, a1);
  k_f2b<<<(B * RNN + 255) / 256, 256, 0, stream>>>(H1b, h1, B * RNN);
  k_f2b<<<(B * RNN + 255) / 256, 256, 0, stream>>>(H2b, h2, B * RNN);
  k_aux32<<<B, AUX, 0, stream>>>(X4, a4);   // X4[:,128:160] = a4

  dim3 blk(256);
  auto gemm = [&](const bf16_t* X, const bf16_t* W, const float* bias,
                  float* Yf, int ldyf, bf16_t* Yb, int ldyb, int N, int K, int relu) {
    dim3 grd(B / 16, N / 128);
    k_wmma_gemm<<<grd, blk, 0, stream>>>(X, W, bias, Yf, ldyf, Yb, ldyb, N, K, relu);
  };

  // i_o = X0 @ I_w^T + I_b  (keep f32 for residual, bf16 for next GEMM)
  gemm(X0, wI, I_b, io_f, RNN, io_b, RNN, RNN, K0, 0);
  // GRU1 gates
  gemm(io_b, w1ih, r1_bih, gi, G3, nullptr, 0, G3, RNN, 0);
  gemm(H1b,  w1hh, r1_bhh, gh, G3, nullptr, 0, G3, RNN, 0);
  // h1' -> h1o ; gru2_in = h1' + i_o -> g2in (f32) and X2[:,0:128] ; X2[:,128:160] = a2
  k_gru_combine<<<B, RNN + AUX, 0, stream>>>(gi, gh, h1, io_f, a2, h1o, g2in, X2);
  // GRU2 gates (gi/gh reused)
  gemm(X2,  w2ih, r2_bih, gi, G3, nullptr, 0, G3, RNN + AUX, 0);
  gemm(H2b, w2hh, r2_bhh, gh, G3, nullptr, 0, G3, RNN, 0);
  // h2' -> h2o ; add1 = gru2_in + h2' -> X3[:,0:128] ; X3[:,128:160] = a3
  k_gru_combine<<<B, RNN + AUX, 0, stream>>>(gi, gh, h2, g2in, a3, h2o, nullptr, X3);
  // relu1 -> X4[:,0:128]
  gemm(X3, wf1, fc1_b, nullptr, 0, X4, FC + AUX, FC, RNN + AUX, 1);
  // relu2 -> R2b
  gemm(X4, wf2, fc2_b, nullptr, 0, R2b, FC, FC, FC + AUX, 1);
  // logits -> d_out
  gemm(R2b, wf3, fc3_b, logits, NC, nullptr, 0, NC, FC, 0);
}
